// NonLocalBlock_23141283791412
// MI455X (gfx1250) — matrix operand order repaired and checked by reading, not compiled, not run on hardware
//
#include <hip/hip_runtime.h>
#include <hip/hip_bf16.h>

// Problem constants (B, C, H, W) = (4, 256, 64, 64)
#define NB  4
#define CC  256
#define CHN 128
#define NN  4096   // H*W

typedef __attribute__((ext_vector_type(16))) _Float16 v16h;
typedef __attribute__((ext_vector_type(8)))  _Float16 v8h;
typedef __attribute__((ext_vector_type(8)))  float    v8f;
typedef __attribute__((ext_vector_type(4)))  int      v4i;

// Async global->LDS path (CDNA5). Guarded so the file compiles on toolchains
// that do not expose the builtins; fallback = synchronous VGPR-bounce copies.
#if __has_builtin(__builtin_amdgcn_global_load_async_to_lds_b128) && \
    __has_builtin(__builtin_amdgcn_s_wait_asynccnt)
#define USE_ASYNC_LDS 1
#else
#define USE_ASYNC_LDS 0
#endif

typedef __attribute__((address_space(1))) v4i* gv4_t;   // global v4i*
typedef __attribute__((address_space(3))) v4i* lv4_t;   // LDS v4i*

__device__ __forceinline__ v8f zero8() {
    v8f r;
#pragma unroll
    for (int i = 0; i < 8; ++i) r[i] = 0.0f;
    return r;
}

// A-fragment (16x32 f16, M=lane%16): elems 0..7 = K(base..+7), 8..15 = K(base+16..+23)
__device__ __forceinline__ v16h load_a16(const _Float16* p) {
    v8h lo = *(const v8h*)p;
    v8h hi = *(const v8h*)(p + 16);
    v16h r;
#pragma unroll
    for (int t = 0; t < 8; ++t) { r[t] = lo[t]; r[t + 8] = hi[t]; }
    return r;
}

// B-fragment (32x16 f16, N=lane%16): elems 0..7 = K(base..+7), 8..15 = K(base+16..+23)
__device__ __forceinline__ v16h load_b16(const _Float16* p) {
    v8h lo = *(const v8h*)p;
    v8h hi = *(const v8h*)(p + 16);
    v16h r;
#pragma unroll
    for (int t = 0; t < 8; ++t) { r[t] = lo[t]; r[t + 8] = hi[t]; }
    return r;
}

// A-fragment sourced from f32 memory with on-the-fly f16 conversion
__device__ __forceinline__ v16h load_a_f32(const float* p) {
    v16h r;
#pragma unroll
    for (int t = 0; t < 8; ++t) {
        r[t]     = (_Float16)p[t];
        r[t + 8] = (_Float16)p[t + 16];
    }
    return r;
}

// Copy 128 contiguous bytes global -> LDS for one lane.
__device__ __forceinline__ void copy128(const _Float16* gsrc, _Float16* ldst) {
#if USE_ASYNC_LDS
    const unsigned long long gb = (unsigned long long)(const void*)gsrc;
    const unsigned lb = (unsigned)(unsigned long long)(void*)ldst;
#pragma unroll
    for (int t = 0; t < 8; ++t)
        __builtin_amdgcn_global_load_async_to_lds_b128(
            (gv4_t)(gb + 16ull * t), (lv4_t)(lb + 16u * t), 0, 0);
#else
#pragma unroll
    for (int t = 0; t < 8; ++t)
        *(v8h*)(ldst + t * 8) = *(const v8h*)(gsrc + t * 8);
#endif
}

__device__ __forceinline__ void wait_async0() {
#if USE_ASYNC_LDS
    __builtin_amdgcn_s_wait_asynccnt(0);
#endif
}

// ---------------------------------------------------------------------------
// Projection: y[b][o][n] = sum_c w[o][c] * x[b][c][n] + bias[o]   (o < CHN)
// Block = 256 threads = 8 waves; block owns a 128-wide n-strip and ALL 128
// o-rows (wave w -> o-tile w). Each 32(c) x 128(n) x-chunk is loaded from
// global exactly ONCE, converted f32->f16 once, staged transposed in LDS,
// and reused by all 8 waves: 64 WMMAs per chunk per block.
// transposed==0: store [b][n][o] f16 ; transposed==1: store [b][o][n] f16
// ---------------------------------------------------------------------------
__global__ __launch_bounds__(256, 1)
void proj_kernel(const float* __restrict__ x, const float* __restrict__ w,
                 const float* __restrict__ bias, _Float16* __restrict__ out,
                 int transposed) {
    constexpr int XP = 40;   // xs pitch (halves): 80 B rows -> spread banks
    __shared__ __attribute__((aligned(16))) _Float16 xs[128 * XP];

    const int tid  = threadIdx.x;
    const int wave = tid >> 5;
    const int lane = tid & 31;
    const int h    = lane >> 4;
    const int lm   = lane & 15;

    const int b  = blockIdx.x >> 5;         // 32 n-strips per batch
    const int n0 = (blockIdx.x & 31) * 128;
    const int o0 = wave * 16;

    // staging role: thread reads 16 consecutive floats of row (c0+cl)
    const int cl = tid >> 3;                // c row 0..31
    const int ng = tid & 7;                 // n group (16 cols each)

    v8f acc[8];
#pragma unroll
    for (int s = 0; s < 8; ++s) acc[s] = zero8();

    const float* wrow = w + (size_t)(o0 + lm) * CC;

    for (int c0 = 0; c0 < CC; c0 += 32) {
        // stage x[c0..c0+32)[n0..n0+128) transposed into xs[n][c] (f16)
        {
            const float* src = x + ((size_t)(b * CC + c0 + cl)) * NN + n0 + ng * 16;
            _Float16* dst = xs + (ng * 16) * XP + cl;
#pragma unroll
            for (int t = 0; t < 16; ++t)
                dst[t * XP] = (_Float16)src[t];
        }
        __syncthreads();

        v16h a = load_a_f32(wrow + c0 + h * 8);
#pragma unroll
        for (int sub = 0; sub < 8; ++sub) {
            v16h bm = load_b16(xs + (sub * 16 + lm) * XP + h * 8);
            acc[sub] = __builtin_amdgcn_wmma_f32_16x16x32_f16(
                false, a, false, bm, (short)0, acc[sub], false, false);
        }
        __syncthreads();
    }

    if (transposed == 0) {
#pragma unroll
        for (int sub = 0; sub < 8; ++sub) {
            _Float16* orow =
                out + ((size_t)(b * NN + n0 + sub * 16 + lm)) * CHN + o0 + h * 8;
#pragma unroll
            for (int r = 0; r < 8; ++r)
                orow[r] = (_Float16)(acc[sub][r] + bias[o0 + h * 8 + r]);
        }
    } else {
#pragma unroll
        for (int sub = 0; sub < 8; ++sub)
#pragma unroll
            for (int r = 0; r < 8; ++r) {
                const int o = o0 + h * 8 + r;
                out[((size_t)(b * CHN + o)) * NN + n0 + sub * 16 + lm] =
                    (_Float16)(acc[sub][r] + bias[o]);
            }
    }
}

// ---------------------------------------------------------------------------
// Flash attention with double-buffered async global->LDS staging.
// O[i][c] = (1/l_i) * sum_j exp(Q_i.K_j - m_i) * V[j][c]
// Block = 4 waves = 64 queries; 64-key tiles; K/V for tile t+1 stream into
// the alternate LDS buffer (ASYNCcnt) while tile t's 32 WMMAs execute.
// ---------------------------------------------------------------------------
__global__ __launch_bounds__(128, 1)
void flash_attn_kernel(const _Float16* __restrict__ thetaT,
                       const _Float16* __restrict__ phiT,
                       const _Float16* __restrict__ gW,
                       _Float16* __restrict__ outT) {
    constexpr int KP = 136;   // kTile pitch (halves)
    constexpr int VP = 72;    // vTile pitch (halves)
    constexpr int PP = 72;    // pScr pitch (halves)
    __shared__ __attribute__((aligned(16))) _Float16 kTile[2][64 * KP];
    __shared__ __attribute__((aligned(16))) _Float16 vTile[2][CHN * VP];
    __shared__ __attribute__((aligned(16))) _Float16 pScr[4][16 * PP];

    const int tid  = threadIdx.x;
    const int wave = tid >> 5;
    const int lane = tid & 31;
    const int h    = lane >> 4;
    const int lm   = lane & 15;
    const int b    = blockIdx.x >> 6;
    const int i0   = (blockIdx.x & 63) * 64 + wave * 16;

    // Q fragments: 4 chunks of K=32 covering ch = 0..127
    v16h qf[4];
    {
        const _Float16* q = thetaT + ((size_t)(b * NN + i0 + lm)) * CHN + h * 8;
#pragma unroll
        for (int c = 0; c < 4; ++c) qf[c] = load_a16(q + c * 32);
    }

    v8f oacc[8];
#pragma unroll
    for (int s = 0; s < 8; ++s) oacc[s] = zero8();
    float rmax[8], rsum[8];
#pragma unroll
    for (int r = 0; r < 8; ++r) { rmax[r] = -1e30f; rsum[r] = 0.0f; }

    // issue staging of one 64-key tile into buffer `buf`
    auto stage = [&](int buf, int j0) {
        const int j = tid >> 1, part = tid & 1;
        copy128(phiT + ((size_t)(b * NN + j0 + j)) * CHN + part * 64,
                &kTile[buf][j * KP + part * 64]);
        copy128(gW + ((size_t)(b * CHN + tid)) * NN + j0,
                &vTile[buf][tid * VP]);
    };

    stage(0, 0);

    constexpr int T = NN / 64;
    for (int t = 0; t < T; ++t) {
        wait_async0();          // this wave's staged tile t has landed
        __syncthreads();        // everyone's tile t landed; buf (t^1) free
        if (t + 1 < T) stage((t + 1) & 1, (t + 1) * 64);   // overlaps compute

        const _Float16* kT = &kTile[t & 1][0];
        const _Float16* vT = &vTile[t & 1][0];

        // S = Q K^T : 16 x 64 per wave
        v8f S[4];
#pragma unroll
        for (int sub = 0; sub < 4; ++sub) {
            v8f s = zero8();
#pragma unroll
            for (int kc = 0; kc < 4; ++kc) {
                v16h bm = load_b16(kT + (sub * 16 + lm) * KP + kc * 32 + h * 8);
                s = __builtin_amdgcn_wmma_f32_16x16x32_f16(
                    false, qf[kc], false, bm, (short)0, s, false, false);
            }
            S[sub] = s;
        }

        // Online softmax. D-layout: VGPR r, lane -> row m = r + 8*h, col = lm.
        float nm[8], alpha[8];
#pragma unroll
        for (int r = 0; r < 8; ++r) {
            float m = S[0][r];
            m = fmaxf(m, S[1][r]);
            m = fmaxf(m, S[2][r]);
            m = fmaxf(m, S[3][r]);
#pragma unroll
            for (int xm = 1; xm < 16; xm <<= 1)
                m = fmaxf(m, __shfl_xor(m, xm, 32));
            nm[r]    = fmaxf(rmax[r], m);
            alpha[r] = __expf(rmax[r] - nm[r]);
        }
#pragma unroll
        for (int r = 0; r < 8; ++r) {
            float ps = 0.0f;
#pragma unroll
            for (int sub = 0; sub < 4; ++sub) {
                float p = __expf(S[sub][r] - nm[r]);
                S[sub][r] = p;
                ps += p;
            }
#pragma unroll
            for (int xm = 1; xm < 16; xm <<= 1) ps += __shfl_xor(ps, xm, 32);
            rsum[r] = rsum[r] * alpha[r] + ps;
            rmax[r] = nm[r];
        }
#pragma unroll
        for (int s = 0; s < 8; ++s)
#pragma unroll
            for (int r = 0; r < 8; ++r) oacc[s][r] *= alpha[r];

        // Bounce P through per-wave LDS scratch: D-layout -> A-layout
        _Float16* pw = pScr[wave];
#pragma unroll
        for (int sub = 0; sub < 4; ++sub)
#pragma unroll
            for (int r = 0; r < 8; ++r)
                pw[(r + 8 * h) * PP + sub * 16 + lm] = (_Float16)S[sub][r];
        asm volatile("s_wait_dscnt 0" ::: "memory");

        // O += P(16x32 over j) x V(32x16 j->ch): 2 j-chunks x 8 c-subtiles
#pragma unroll
        for (int kc = 0; kc < 2; ++kc) {
            v16h pa = load_a16(pw + lm * PP + kc * 32 + h * 8);
#pragma unroll
            for (int cs = 0; cs < 8; ++cs) {
                v16h vb = load_b16(vT + (cs * 16 + lm) * VP + kc * 32 + h * 8);
                oacc[cs] = __builtin_amdgcn_wmma_f32_16x16x32_f16(
                    false, pa, false, vb, (short)0, oacc[cs], false, false);
            }
        }
    }

    // Epilogue: normalize by l and store O as [b][n][ch] f16
#pragma unroll
    for (int cs = 0; cs < 8; ++cs)
#pragma unroll
        for (int r = 0; r < 8; ++r) {
            const int i = i0 + r + 8 * h;
            const float v = oacc[cs][r] / rsum[r];
            outT[((size_t)(b * NN + i)) * CHN + cs * 16 + lm] = (_Float16)v;
        }
}

// ---------------------------------------------------------------------------
// Final: out[b][o][n] = sum_ch Ww[o][ch]*O[b][n][ch] + Wb[o] + x[b][o][n]
// One wave = one 16(o) x 16(n) tile, K-loop over CH=128 -> 4 WMMAs.
// ---------------------------------------------------------------------------
__global__ __launch_bounds__(256, 1)
void final_proj_kernel(const _Float16* __restrict__ OT,
                       const float* __restrict__ Ww,
                       const float* __restrict__ Wb,
                       const float* __restrict__ x,
                       float* __restrict__ out) {
    const int tid  = threadIdx.x;
    const int wave = tid >> 5;
    const int lane = tid & 31;
    const int h    = lane >> 4;
    const int lm   = lane & 15;

    int tile = blockIdx.x * 8 + wave;      // 4 * 16 * 256 = 16384 tiles
    const int b  = tile >> 12;
    tile &= 4095;
    const int o0 = (tile >> 8) * 16;
    const int n0 = (tile & 255) * 16;

    v8f acc = zero8();
    const float* wrow = Ww + (size_t)(o0 + lm) * CHN;
#pragma unroll
    for (int c0 = 0; c0 < CHN; c0 += 32) {
        v16h a  = load_a_f32(wrow + c0 + h * 8);
        v16h bm = load_b16(OT + ((size_t)(b * NN + n0 + lm)) * CHN + c0 + h * 8);
        acc = __builtin_amdgcn_wmma_f32_16x16x32_f16(false, a, false, bm,
                                                     (short)0, acc, false, false);
    }
#pragma unroll
    for (int r = 0; r < 8; ++r) {
        const int o = o0 + h * 8 + r;
        const size_t idx = ((size_t)(b * CC + o)) * NN + n0 + lm;
        out[idx] = acc[r] + Wb[o] + x[idx];
    }
}

// ---------------------------------------------------------------------------
extern "C" void kernel_launch(void* const* d_in, const int* in_sizes, int n_in,
                              void* d_out, int out_size, void* d_ws,
                              size_t ws_size, hipStream_t stream) {
    (void)in_sizes; (void)n_in; (void)out_size; (void)ws_size;

    const float* x       = (const float*)d_in[0];
    const float* theta_w = (const float*)d_in[1];
    const float* theta_b = (const float*)d_in[2];
    const float* phi_w   = (const float*)d_in[3];
    const float* phi_b   = (const float*)d_in[4];
    const float* g_w     = (const float*)d_in[5];
    const float* g_b     = (const float*)d_in[6];
    const float* W_w     = (const float*)d_in[7];
    const float* W_b     = (const float*)d_in[8];
    float* out = (float*)d_out;

    const size_t SEG = (size_t)NB * NN * CHN;
    _Float16* thetaT = (_Float16*)d_ws;            // [b][n][ch]
    _Float16* phiT   = thetaT + SEG;               // [b][n][ch]
    _Float16* gWv    = phiT + SEG;                 // [b][ch][n]
    _Float16* outT   = gWv + SEG;                  // [b][n][ch]

    proj_kernel<<<NB * 32, 256, 0, stream>>>(x, theta_w, theta_b, thetaT, 0);
    proj_kernel<<<NB * 32, 256, 0, stream>>>(x, phi_w,   phi_b,   phiT,   0);
    proj_kernel<<<NB * 32, 256, 0, stream>>>(x, g_w,     g_b,     gWv,    1);
    flash_attn_kernel<<<NB * (NN / 64), 128, 0, stream>>>(thetaT, phiT, gWv, outT);
    final_proj_kernel<<<2048, 256, 0, stream>>>(outT, W_w, W_b, x, out);
}
